// GINet_82884278878970
// MI455X (gfx1250) — matrix operand order repaired and checked by reading, not compiled, not run on hardware
//
#include <hip/hip_runtime.h>
#include <hip/hip_bf16.h>

// ---------------------------------------------------------------------------
// Problem constants (from reference)
// ---------------------------------------------------------------------------
#define N_NODES   50000
#define N_EDGES   150000
#define NB        1024
#define EMB       300
#define FEAT      512
#define NLAYERS   5
#define FP_BITS   2215
#define FPH       512
#define FUSED_DIM 1024

typedef __attribute__((ext_vector_type(16))) _Float16 v16h;
typedef __attribute__((ext_vector_type(2)))  _Float16 v2h;
typedef __attribute__((ext_vector_type(8)))  float    v8f;

// ---------------------------------------------------------------------------
// Generic WMMA GEMM:  C[M,N] = act(A[M,K] @ W[K,N] + bias[N])
// act: 0 = none, 1 = relu, 2 = sigmoid
// Block: 256 threads (8 wave32), tile 64(M) x 128(N), K-step 32.
// Each wave computes a 32x32 sub-tile = 2x2 WMMA 16x16 accumulators.
//
// LDS tiles are FRAGMENT-MAJOR: [tile16][lane][16 halves] -> each lane's
// v16h fragment is 32 contiguous bytes (2x ds_load_b128).
//
// Wave32 WMMA f16 layouts (ISA 7.12.2):
//   A (16x32): lane L -> M=L%16, grp=L/16; half h -> K=((h<8)?h:h+8)+8*grp
//     inverse (even pairs): K 0..7 -> (g0,h=K); 8..15 -> (g1,h=K-8);
//                           16..23 -> (g0,h=K-8); 24..31 -> (g1,h=K-16)
//   B (32x16): lane L -> N=L%16, grp=L/16; half h -> K=((h<8)?h:h+8)+8*grp (same as A)
//   C/D (16x16 f32): VGPR r -> M=r+8*(L/16), N=L%16
//
// Interior blocks take a guard-free fast path (uniform scalar branch), so
// the per-element exec-mask dance only runs on boundary tiles.
// ---------------------------------------------------------------------------
#define TILE_M 64
#define TILE_N 128
#define TILE_K 32

__global__ __launch_bounds__(256) void gemm_bias_act(
    const float* __restrict__ A, const float* __restrict__ W,
    const float* __restrict__ bias, float* __restrict__ C,
    int M, int K, int N, int act)
{
    __shared__ __align__(32) _Float16 As[(TILE_M / 16) * 32 * 16];  // 4KB
    __shared__ __align__(32) _Float16 Bs[(TILE_N / 16) * 32 * 16];  // 8KB

    const int tid  = threadIdx.x;
    const int wave = tid >> 5;
    const int lane = tid & 31;
    const int block_m = blockIdx.y * TILE_M;
    const int block_n = blockIdx.x * TILE_N;
    const int wm = (wave >> 2) * 32;   // 0 / 32
    const int wn = (wave & 3) * 32;    // 0..96
    const int lrow = lane & 15;
    const int lgrp = lane >> 4;

    const bool fullM = (block_m + TILE_M <= M);
    const bool fullN = (block_n + TILE_N <= N);

    v8f c00 = {}, c01 = {}, c10 = {}, c11 = {};

    for (int k0 = 0; k0 < K; k0 += TILE_K) {
        const bool fullK = (k0 + TILE_K <= K);

        // ---- Stage A tile: 64 rows x 16 k-pairs = 1024 slots, 4/thread ----
        if (fullM && fullK) {
            // interior fast path: no per-element guards
            if (k0 + 2 * TILE_K <= K) {  // prefetch next K-panel of A stream
                __builtin_prefetch(&A[(long)(block_m + (tid >> 2)) * K + k0 + TILE_K + ((tid & 3) << 3)], 0, 0);
            }
#pragma unroll
            for (int i = tid; i < TILE_M * (TILE_K / 2); i += 256) {
                int m  = i >> 4;
                int kl = (i & 15) << 1;
                const float* ap = &A[(long)(block_m + m) * K + k0 + kl];
                int blk = kl >> 3;
                int g   = blk & 1;
                int h   = kl - (((blk + 1) >> 1) << 3);
                int slot = ((m >> 4) * 32 + g * 16 + (m & 15)) * 16 + h;
                v2h pk; pk[0] = (_Float16)ap[0]; pk[1] = (_Float16)ap[1];
                *(v2h*)&As[slot] = pk;
            }
        } else {
            for (int i = tid; i < TILE_M * (TILE_K / 2); i += 256) {
                int m  = i >> 4;
                int kl = (i & 15) << 1;
                int gm = block_m + m, gk = k0 + kl;
                float f0 = (gm < M && gk     < K) ? A[(long)gm * K + gk]     : 0.0f;
                float f1 = (gm < M && gk + 1 < K) ? A[(long)gm * K + gk + 1] : 0.0f;
                int blk = kl >> 3;
                int g   = blk & 1;
                int h   = kl - (((blk + 1) >> 1) << 3);
                int slot = ((m >> 4) * 32 + g * 16 + (m & 15)) * 16 + h;
                v2h pk; pk[0] = (_Float16)f0; pk[1] = (_Float16)f1;
                *(v2h*)&As[slot] = pk;
            }
        }

        // ---- Stage B tile: 128 cols x 16 k-pairs = 2048 slots, 8/thread ----
        if (fullN && fullK) {
#pragma unroll
            for (int i = tid; i < TILE_N * (TILE_K / 2); i += 256) {
                int n  = i >> 4;
                int kl = (i & 15) << 1;
                const float* wp = &W[(long)(k0 + kl) * N + block_n + n];
                int g = (kl >> 3) & 1;
                int h = kl - ((((kl >> 3) + 1) >> 1) << 3);
                int slot = ((n >> 4) * 32 + g * 16 + (n & 15)) * 16 + h;
                v2h pk; pk[0] = (_Float16)wp[0]; pk[1] = (_Float16)wp[N];
                *(v2h*)&Bs[slot] = pk;
            }
        } else {
            for (int i = tid; i < TILE_N * (TILE_K / 2); i += 256) {
                int n  = i >> 4;
                int kl = (i & 15) << 1;
                int gk = k0 + kl, gn = block_n + n;
                float f0 = (gk     < K && gn < N) ? W[(long)gk * N + gn]       : 0.0f;
                float f1 = (gk + 1 < K && gn < N) ? W[(long)(gk + 1) * N + gn] : 0.0f;
                int g = (kl >> 3) & 1;
                int h = kl - ((((kl >> 3) + 1) >> 1) << 3);
                int slot = ((n >> 4) * 32 + g * 16 + (n & 15)) * 16 + h;
                v2h pk; pk[0] = (_Float16)f0; pk[1] = (_Float16)f1;
                *(v2h*)&Bs[slot] = pk;
            }
        }
        __syncthreads();

        // ---- Fragment loads: 32B contiguous per lane (2x ds_load_b128) ----
        v16h a0 = *(const v16h*)&As[(((wm >> 4) + 0) * 32 + lane) * 16];
        v16h a1 = *(const v16h*)&As[(((wm >> 4) + 1) * 32 + lane) * 16];
        v16h b0 = *(const v16h*)&Bs[(((wn >> 4) + 0) * 32 + lane) * 16];
        v16h b1 = *(const v16h*)&Bs[(((wn >> 4) + 1) * 32 + lane) * 16];

        c00 = __builtin_amdgcn_wmma_f32_16x16x32_f16(false, a0, false, b0, (short)0, c00, false, false);
        c01 = __builtin_amdgcn_wmma_f32_16x16x32_f16(false, a0, false, b1, (short)0, c01, false, false);
        c10 = __builtin_amdgcn_wmma_f32_16x16x32_f16(false, a1, false, b0, (short)0, c10, false, false);
        c11 = __builtin_amdgcn_wmma_f32_16x16x32_f16(false, a1, false, b1, (short)0, c11, false, false);
        __syncthreads();
    }

    // ---- Epilogue: M = r + 8*(lane/16), N = lane%16 within each 16x16 ----
    // guard=0 variant is a uniform specialization for interior C tiles.
    auto store_tile = [&](const v8f& acc, int tm, int tn, int guard) {
#pragma unroll
        for (int r = 0; r < 8; ++r) {
            int gm = block_m + wm + tm + lgrp * 8 + r;
            int gn = block_n + wn + tn + lrow;
            if (!guard || (gm < M && gn < N)) {
                float v = acc[r] + bias[gn];
                if (act == 1)      v = fmaxf(v, 0.0f);
                else if (act == 2) v = 1.0f / (1.0f + __expf(-v));
                C[(long)gm * N + gn] = v;
            }
        }
    };
    if (fullM && fullN) {
        store_tile(c00,  0,  0, 0);
        store_tile(c01,  0, 16, 0);
        store_tile(c10, 16,  0, 0);
        store_tile(c11, 16, 16, 0);
    } else {
        store_tile(c00,  0,  0, 1);
        store_tile(c01,  0, 16, 1);
        store_tile(c10, 16,  0, 1);
        store_tile(c11, 16, 16, 1);
    }
}

// ---------------------------------------------------------------------------
// Elementwise / scatter kernels (bandwidth-bound phases)
// ---------------------------------------------------------------------------
__global__ void k_fill_zero(float* p, long n) {
    long i = (long)blockIdx.x * blockDim.x + threadIdx.x;
    if (i < n) p[i] = 0.0f;
}

// h[i,c] = x_emb1[x[i,0],c] + x_emb2[x[i,1],c]
__global__ void k_embed(const int* __restrict__ x, const float* __restrict__ e1,
                        const float* __restrict__ e2, float* __restrict__ h,
                        int n, int C) {
    long i = (long)blockIdx.x * blockDim.x + threadIdx.x;
    if (i >= (long)n * C) return;
    int node = (int)(i / C), c = (int)(i % C);
    h[i] = e1[(long)x[node * 2 + 0] * C + c] + e2[(long)x[node * 2 + 1] * C + c];
}

// self-loop: aggr[i,c] = h[i,c] + ee1[4,c] + ee2[0,c]
__global__ void k_msg_self(const float* __restrict__ h, const float* __restrict__ ee1,
                           const float* __restrict__ ee2, float* __restrict__ aggr,
                           int n, int C) {
    long i = (long)blockIdx.x * blockDim.x + threadIdx.x;
    if (i >= (long)n * C) return;
    int c = (int)(i % C);
    aggr[i] = h[i] + ee1[4 * C + c] + ee2[c];
}

// real edges: aggr[dst,c] += h[src,c] + ee1[ea0,c] + ee2[ea1,c]
__global__ void k_msg_scatter(const float* __restrict__ h, const int* __restrict__ src,
                              const int* __restrict__ dst, const int* __restrict__ eattr,
                              const float* __restrict__ ee1, const float* __restrict__ ee2,
                              float* __restrict__ aggr, int E, int C) {
    long i = (long)blockIdx.x * blockDim.x + threadIdx.x;
    if (i >= (long)E * C) return;
    int e = (int)(i / C), c = (int)(i % C);
    int a0 = eattr[e * 2 + 0], a1 = eattr[e * 2 + 1];
    float m = h[(long)src[e] * C + c] + ee1[(long)a0 * C + c] + ee2[(long)a1 * C + c];
    atomicAdd(&aggr[(long)dst[e] * C + c], m);
}

// per-channel mean/var over n rows (one block per channel)
__global__ void k_bn_stats(const float* __restrict__ h, float* __restrict__ mean,
                           float* __restrict__ var, int n, int C) {
    int c = blockIdx.x;
    float s = 0.0f, s2 = 0.0f;
    for (int i = threadIdx.x; i < n; i += blockDim.x) {
        float v = h[(long)i * C + c];
        s += v; s2 += v * v;
    }
    __shared__ float sh[256], sh2[256];
    sh[threadIdx.x] = s; sh2[threadIdx.x] = s2;
    __syncthreads();
    for (int off = 128; off > 0; off >>= 1) {
        if (threadIdx.x < off) {
            sh[threadIdx.x]  += sh[threadIdx.x + off];
            sh2[threadIdx.x] += sh2[threadIdx.x + off];
        }
        __syncthreads();
    }
    if (threadIdx.x == 0) {
        float m = sh[0] / (float)n;
        mean[c] = m;
        var[c]  = sh2[0] / (float)n - m * m;
    }
}

__global__ void k_bn_apply(float* __restrict__ h, const float* __restrict__ mean,
                           const float* __restrict__ var, const float* __restrict__ g,
                           const float* __restrict__ b, int n, int C, int do_relu) {
    long i = (long)blockIdx.x * blockDim.x + threadIdx.x;
    if (i >= (long)n * C) return;
    int c = (int)(i % C);
    float v = (h[i] - mean[c]) * rsqrtf(var[c] + 1e-5f) * g[c] + b[c];
    h[i] = do_relu ? fmaxf(v, 0.0f) : v;
}

// one wave per node: gate = sigmoid(dot(h[node,:], w) + b)
__global__ void k_gate(const float* __restrict__ h, const float* __restrict__ w,
                       const float* __restrict__ b, float* __restrict__ gate,
                       float* __restrict__ attn, int n, int C) {
    int node = blockIdx.x * 8 + (threadIdx.x >> 5);
    int lane = threadIdx.x & 31;
    if (node >= n) return;
    float s = 0.0f;
    for (int c = lane; c < C; c += 32) s += h[(long)node * C + c] * w[c];
#pragma unroll
    for (int off = 16; off > 0; off >>= 1) s += __shfl_down(s, off, 32);
    if (lane == 0) {
        float g = 1.0f / (1.0f + __expf(-(s + b[0])));
        gate[node] = g;
        attn[node] = g;
    }
}

// gate in (0,1) -> positive-float ordering == int ordering of bit pattern
__global__ void k_segmax(const float* __restrict__ gate, const int* __restrict__ batch,
                         int* __restrict__ gmax_bits, int n) {
    int i = blockIdx.x * blockDim.x + threadIdx.x;
    if (i >= n) return;
    atomicMax(&gmax_bits[batch[i]], __float_as_int(gate[i]));
}

__global__ void k_expsum(const float* __restrict__ gate, const int* __restrict__ batch,
                         const int* __restrict__ gmax_bits, float* __restrict__ ge,
                         float* __restrict__ gsum, int n) {
    int i = blockIdx.x * blockDim.x + threadIdx.x;
    if (i >= n) return;
    float e = __expf(gate[i] - __int_as_float(gmax_bits[batch[i]]));
    ge[i] = e;
    atomicAdd(&gsum[batch[i]], e);
}

// hg[batch[i],c] += (ge[i]/gsum[batch[i]]) * h[i,c]
__global__ void k_pool(const float* __restrict__ h, const float* __restrict__ ge,
                       const float* __restrict__ gsum, const int* __restrict__ batch,
                       float* __restrict__ hg, int n, int C) {
    long i = (long)blockIdx.x * blockDim.x + threadIdx.x;
    if (i >= (long)n * C) return;
    int node = (int)(i / C), c = (int)(i % C);
    int b = batch[node];
    float coef = ge[node] / gsum[b];
    atomicAdd(&hg[(long)b * C + c], coef * h[i]);
}

__global__ void k_concat(const float* __restrict__ a, const float* __restrict__ b,
                         float* __restrict__ z, int rows, int half) {
    long i = (long)blockIdx.x * blockDim.x + threadIdx.x;
    if (i >= (long)rows * 2 * half) return;
    int r = (int)(i / (2 * half)), j = (int)(i % (2 * half));
    z[i] = (j < half) ? a[(long)r * half + j] : b[(long)r * half + (j - half)];
}

// fused = v + g*(g*v - v), split halves (hG/gh | fpm/gfp)
__global__ void k_fuse(const float* __restrict__ hG, const float* __restrict__ fpm,
                       const float* __restrict__ gh, const float* __restrict__ gfp,
                       float* __restrict__ fused, int rows, int half) {
    long i = (long)blockIdx.x * blockDim.x + threadIdx.x;
    if (i >= (long)rows * 2 * half) return;
    int r = (int)(i / (2 * half)), j = (int)(i % (2 * half));
    float v, g;
    if (j < half) { v = hG[(long)r * half + j];           g = gh[(long)r * half + j]; }
    else          { v = fpm[(long)r * half + (j - half)]; g = gfp[(long)r * half + (j - half)]; }
    fused[i] = v + g * (g * v - v);
}

// ---------------------------------------------------------------------------
// Host orchestration
// ---------------------------------------------------------------------------
static inline unsigned nblk(long n) { return (unsigned)((n + 255) / 256); }

extern "C" void kernel_launch(void* const* d_in, const int* in_sizes, int n_in,
                              void* d_out, int out_size, void* d_ws, size_t ws_size,
                              hipStream_t stream) {
    (void)in_sizes; (void)n_in; (void)out_size; (void)ws_size;

    const int*   x          = (const int*)d_in[0];
    const int*   edge_index = (const int*)d_in[1];
    const int*   edge_attr  = (const int*)d_in[2];
    const int*   batch      = (const int*)d_in[3];
    const float* fp         = (const float*)d_in[4];
    const float* x_emb1     = (const float*)d_in[5];
    const float* x_emb2     = (const float*)d_in[6];
    const float* edge_emb1  = (const float*)d_in[7];
    const float* edge_emb2  = (const float*)d_in[8];
    const float* mlp_w1     = (const float*)d_in[9];
    const float* mlp_b1     = (const float*)d_in[10];
    const float* mlp_w2     = (const float*)d_in[11];
    const float* mlp_b2     = (const float*)d_in[12];
    const float* bn_g       = (const float*)d_in[13];
    const float* bn_b       = (const float*)d_in[14];
    const float* feat_w     = (const float*)d_in[15];
    const float* feat_b     = (const float*)d_in[16];
    const float* gate_nn_w  = (const float*)d_in[17];
    const float* gate_nn_b  = (const float*)d_in[18];
    const float* fp_w[5]    = {(const float*)d_in[19], (const float*)d_in[20], (const float*)d_in[21],
                               (const float*)d_in[22], (const float*)d_in[23]};
    const float* fp_b[5]    = {(const float*)d_in[24], (const float*)d_in[25], (const float*)d_in[26],
                               (const float*)d_in[27], (const float*)d_in[28]};
    const float* gate_h_w   = (const float*)d_in[29];
    const float* gate_h_b   = (const float*)d_in[30];
    const float* gate_fp_w  = (const float*)d_in[31];
    const float* gate_fp_b  = (const float*)d_in[32];
    const float* pred_w[3]  = {(const float*)d_in[33], (const float*)d_in[34], (const float*)d_in[35]};
    const float* pred_b[3]  = {(const float*)d_in[36], (const float*)d_in[37], (const float*)d_in[38]};

    // Workspace carve-out
    char* ws = (char*)d_ws;
    size_t off = 0;
    auto carve = [&](size_t bytes) -> float* {
        float* p = (float*)(ws + off);
        off = (off + bytes + 255) & ~(size_t)255;
        return p;
    };
    float* h      = carve((size_t)N_NODES * EMB * 4);
    float* aggr   = carve((size_t)N_NODES * EMB * 4);
    float* tmp600 = carve((size_t)N_NODES * 2 * EMB * 4);
    float* gatev  = carve((size_t)N_NODES * 4);
    float* ge     = carve((size_t)N_NODES * 4);
    float* gmax   = carve((size_t)NB * 4);          // reinterpreted as int bits
    float* gsum   = carve((size_t)NB * 4);
    float* meanb  = carve((size_t)EMB * 4);
    float* varb   = carve((size_t)EMB * 4);
    float* hg     = carve((size_t)NB * EMB * 4);
    float* fpa    = carve((size_t)NB * 2048 * 4);
    float* fpb    = carve((size_t)NB * 2048 * 4);
    float* z      = carve((size_t)NB * FUSED_DIM * 4);
    float* ghb    = carve((size_t)NB * FEAT * 4);
    float* gfpb   = carve((size_t)NB * FPH * 4);
    float* fused  = carve((size_t)NB * FUSED_DIM * 4);
    float* p1     = carve((size_t)NB * 512 * 4);
    float* p2     = carve((size_t)NB * 512 * 4);

    float* hG       = (float*)d_out;                     // [1024,512]
    float* o_out    = (float*)d_out + (size_t)NB * FEAT; // [1024,2]
    float* attn_out = o_out + (size_t)NB * 2;            // [50000]

    auto gemm = [&](const float* A, const float* W, const float* b, float* C,
                    int M, int K, int N, int act) {
        dim3 grid((unsigned)((N + TILE_N - 1) / TILE_N), (unsigned)((M + TILE_M - 1) / TILE_M));
        gemm_bias_act<<<grid, dim3(256), 0, stream>>>(A, W, b, C, M, K, N, act);
    };

    // ---- node embedding ----
    k_embed<<<nblk((long)N_NODES * EMB), 256, 0, stream>>>(x, x_emb1, x_emb2, h, N_NODES, EMB);

    const int* e_src = edge_index;
    const int* e_dst = edge_index + N_EDGES;

    // ---- 5 GNN layers ----
    for (int l = 0; l < NLAYERS; ++l) {
        const float* ee1 = edge_emb1 + (size_t)l * 5 * EMB;
        const float* ee2 = edge_emb2 + (size_t)l * 3 * EMB;
        k_msg_self<<<nblk((long)N_NODES * EMB), 256, 0, stream>>>(h, ee1, ee2, aggr, N_NODES, EMB);
        k_msg_scatter<<<nblk((long)N_EDGES * EMB), 256, 0, stream>>>(h, e_src, e_dst, edge_attr,
                                                                     ee1, ee2, aggr, N_EDGES, EMB);
        gemm(aggr,   mlp_w1 + (size_t)l * EMB * 2 * EMB, mlp_b1 + (size_t)l * 2 * EMB,
             tmp600, N_NODES, EMB, 2 * EMB, /*relu*/1);
        gemm(tmp600, mlp_w2 + (size_t)l * 2 * EMB * EMB, mlp_b2 + (size_t)l * EMB,
             h,      N_NODES, 2 * EMB, EMB, /*none*/0);
        k_bn_stats<<<EMB, 256, 0, stream>>>(h, meanb, varb, N_NODES, EMB);
        k_bn_apply<<<nblk((long)N_NODES * EMB), 256, 0, stream>>>(h, meanb, varb,
            bn_g + (size_t)l * EMB, bn_b + (size_t)l * EMB, N_NODES, EMB, (l < NLAYERS - 1) ? 1 : 0);
    }

    // ---- gated attention pooling ----
    k_gate<<<(unsigned)((N_NODES + 7) / 8), 256, 0, stream>>>(h, gate_nn_w, gate_nn_b,
                                                              gatev, attn_out, N_NODES, EMB);
    k_fill_zero<<<nblk(NB), 256, 0, stream>>>(gmax, NB);
    k_fill_zero<<<nblk(NB), 256, 0, stream>>>(gsum, NB);
    k_fill_zero<<<nblk((long)NB * EMB), 256, 0, stream>>>(hg, (long)NB * EMB);
    k_segmax<<<nblk(N_NODES), 256, 0, stream>>>(gatev, batch, (int*)gmax, N_NODES);
    k_expsum<<<nblk(N_NODES), 256, 0, stream>>>(gatev, batch, (const int*)gmax, ge, gsum, N_NODES);
    k_pool<<<nblk((long)N_NODES * EMB), 256, 0, stream>>>(h, ge, gsum, batch, hg, N_NODES, EMB);
    gemm(hg, feat_w, feat_b, hG, NB, EMB, FEAT, 0);      // hG -> d_out directly

    // ---- fingerprint MLP ----
    gemm(fp,  fp_w[0], fp_b[0], fpa, NB, FP_BITS, 2048, 1);
    gemm(fpa, fp_w[1], fp_b[1], fpb, NB, 2048,    1024, 1);
    gemm(fpb, fp_w[2], fp_b[2], fpa, NB, 1024,    1024, 1);
    gemm(fpa, fp_w[3], fp_b[3], fpb, NB, 1024,     512, 1);
    gemm(fpb, fp_w[4], fp_b[4], fpa, NB, 512,      FPH, 1);
    float* fpm = fpa;

    // ---- fusion gates + prediction head ----
    k_concat<<<nblk((long)NB * FUSED_DIM), 256, 0, stream>>>(hG, fpm, z, NB, FEAT);
    gemm(z, gate_h_w,  gate_h_b,  ghb,  NB, FUSED_DIM, FEAT, /*sigmoid*/2);
    gemm(z, gate_fp_w, gate_fp_b, gfpb, NB, FUSED_DIM, FPH,  /*sigmoid*/2);
    k_fuse<<<nblk((long)NB * FUSED_DIM), 256, 0, stream>>>(hG, fpm, ghb, gfpb, fused, NB, FEAT);
    gemm(fused, pred_w[0], pred_b[0], p1,    NB, FUSED_DIM, 512, 1);
    gemm(p1,    pred_w[1], pred_b[1], p2,    NB, 512,       512, 1);
    gemm(p2,    pred_w[2], pred_b[2], o_out, NB, 512,         2, 0);
}
